// RNNModule_21706764714107
// MI455X (gfx1250) — matrix operand order repaired and checked by reading, not compiled, not run on hardware
//
#include <hip/hip_runtime.h>

#define NN      1024
#define NBATCH  256
#define TSTEPS  300
#define ALPHA_F     0.1f
#define NOISE_AMP_F 0.03f

typedef __attribute__((ext_vector_type(16))) __bf16        v16bf;
typedef __attribute__((ext_vector_type(8)))  float         v8f;
typedef __attribute__((ext_vector_type(4)))  unsigned int  v4u;

union Frag { v16bf v; v4u q[2]; };

__device__ __forceinline__ unsigned short f32_to_bf16_rne(float f) {
    unsigned int u = __float_as_uint(f);
    u += 0x7FFFu + ((u >> 16) & 1u);   // round-to-nearest-even
    return (unsigned short)(u >> 16);
}

// ---------------------------------------------------------------------------
// Zero the f32 state x, the t=0 bf16 rate buffer, and out[0] (= tanh(0) = 0).
// ---------------------------------------------------------------------------
__global__ void rnn_init(float* __restrict__ x,
                         unsigned short* __restrict__ rbf0,
                         float* __restrict__ out0) {
    int i = blockIdx.x * blockDim.x + threadIdx.x;
    if (i < NBATCH * NN) {
        x[i]    = 0.0f;
        rbf0[i] = 0;
        out0[i] = 0.0f;
    }
}

// ---------------------------------------------------------------------------
// W_bf[n][k] = bf16( mask[n][k] ? W[n][k] : 0 ), row-major N x K.
// Row n of W_eff is WMMA B-matrix column n -> two 16B per-lane chunks, 32B apart.
// ---------------------------------------------------------------------------
__global__ void rnn_mask_w(const float* __restrict__ W,
                           const unsigned char* __restrict__ mask,
                           unsigned short* __restrict__ wbf) {
    int i = blockIdx.x * blockDim.x + threadIdx.x;
    float w = mask[i] ? W[i] : 0.0f;
    wbf[i] = f32_to_bf16_rne(w);
}

// ---------------------------------------------------------------------------
// One Euler step: acc = r @ W_eff^T (bf16 WMMA, f32 accum), then
//   x += alpha*(-x + acc + inp + bias) + alpha*noise_amp*nz ; r = tanh(x)
//
// Block = 128 threads (4 wave32). blockIdx.y -> 32 batch rows (2 M-tiles,
// staged in LDS via async global->LDS). blockIdx.x -> 128 columns; each wave
// owns 2 adjacent N-tiles. 2Mx2N register blocking: 4 wmma per k-step from
// 2 A-frags (LDS) + 2 B-frags (global), 4 independent accumulator chains.
// ---------------------------------------------------------------------------
__global__ __launch_bounds__(128)
void rnn_step(const unsigned short* __restrict__ rbf_in,
              const unsigned short* __restrict__ wbf,
              const float* __restrict__ inp_t,
              const float* __restrict__ nz_t,
              const float* __restrict__ bias,
              float* __restrict__ x,
              float* __restrict__ out_t1,
              unsigned short* __restrict__ rbf_out) {
    // 32 rows x 1024 bf16 = 64 KB A-slab (contiguous slab of rbf_in)
    __shared__ alignas(16) unsigned char sA[32 * NN * 2];

    const int tid    = threadIdx.x;
    const int lane   = tid & 31;
    const int wave   = tid >> 5;
    const int msup   = blockIdx.y;            // 32-row super tile
    const int ntile0 = blockIdx.x * 8 + wave * 2;

    // --- Async global->LDS staging of the 64KB r-slab (CDNA5 path) ---------
    // 128 threads x 32 chunks x 16B. ASYNCcnt tracked; generic->LDS pointers
    // carry the LDS byte offset in their low 32 bits.
    {
        const unsigned char* gsrc =
            (const unsigned char*)(rbf_in + (size_t)msup * 32 * NN) + tid * 16;
        unsigned ldsAddr = (unsigned)(unsigned long long)(void*)(sA + tid * 16);
#pragma unroll
        for (int i = 0; i < 32; ++i) {
            asm volatile("global_load_async_to_lds_b128 %0, %1, off offset:%2"
                         :: "v"(ldsAddr), "v"(gsrc), "i"(i * 2048)
                         : "memory");
        }
        asm volatile("s_wait_asynccnt 0x0" ::: "memory");
    }
    __syncthreads();

    // A frag (16-bit A 16x32 layout): lane<16 -> row=lane,    K = kb+{0..7,16..23}
    //                                 lane>=16 -> row=lane-16, K = kb+{8..15,24..31}
    const int rowA  = lane & 15;
    const int koff8 = (lane >> 4) * 8;
    const unsigned char* aBase0 = sA + rowA * (NN * 2) + koff8 * 2;
    const unsigned char* aBase1 = aBase0 + 16 * (NN * 2);

    // B frag (32x16): lane<16 -> N=lane, K = kb+{0..7,16..23} ; lane>=16 -> K = kb+{8..15,24..31}
    const int nloc = lane & 15;
    const int n0   = ntile0 * 16 + nloc;
    const unsigned short* __restrict__ wrow0 =
        wbf + (size_t)n0 * NN + (lane >> 4) * 8;
    const unsigned short* __restrict__ wrow1 = wrow0 + 16 * NN;

    v8f acc00 = {}, acc01 = {}, acc10 = {}, acc11 = {};
    for (int kb = 0; kb < NN; kb += 32) {
        Frag a0, a1, b0, b1;
        const v4u* ap0 = (const v4u*)(aBase0 + kb * 2);
        const v4u* ap1 = (const v4u*)(aBase1 + kb * 2);
        a0.q[0] = ap0[0];  a0.q[1] = ap0[2];   // +32B = K+16 half
        a1.q[0] = ap1[0];  a1.q[1] = ap1[2];
        const v4u* bp0 = (const v4u*)(wrow0 + kb);
        const v4u* bp1 = (const v4u*)(wrow1 + kb);
        b0.q[0] = bp0[0];  b0.q[1] = bp0[2];
        b1.q[0] = bp1[0];  b1.q[1] = bp1[2];
        acc00 = __builtin_amdgcn_wmma_f32_16x16x32_bf16(false, a0.v, false, b0.v,
                                                        (short)0, acc00, false, false);
        acc01 = __builtin_amdgcn_wmma_f32_16x16x32_bf16(false, a0.v, false, b1.v,
                                                        (short)0, acc01, false, false);
        acc10 = __builtin_amdgcn_wmma_f32_16x16x32_bf16(false, a1.v, false, b0.v,
                                                        (short)0, acc10, false, false);
        acc11 = __builtin_amdgcn_wmma_f32_16x16x32_bf16(false, a1.v, false, b1.v,
                                                        (short)0, acc11, false, false);
    }

    // Epilogue. C/D layout: lane<16: M=v, N=lane ; lane>=16: M=v+8, N=lane-16
    const float bn[2] = { bias[n0], bias[n0 + 16] };
    const int vbase   = (lane >> 4) ? 8 : 0;
    const v8f* accs[2][2] = { { &acc00, &acc01 }, { &acc10, &acc11 } };
#pragma unroll
    for (int mt = 0; mt < 2; ++mt) {
#pragma unroll
        for (int nt = 0; nt < 2; ++nt) {
            const v8f a = *accs[mt][nt];
            const int n = n0 + nt * 16;
#pragma unroll
            for (int v = 0; v < 8; ++v) {
                const int    b   = msup * 32 + mt * 16 + vbase + v;
                const size_t idx = (size_t)b * NN + n;
                const float xo = x[idx];
                const float xi = inp_t[idx];
                const float nz = nz_t[idx];
                const float xn = xo + ALPHA_F * (-xo + a[v] + xi + bn[nt])
                               + (ALPHA_F * NOISE_AMP_F) * nz;
                const float rn = tanhf(xn);
                x[idx]       = xn;
                out_t1[idx]  = rn;
                rbf_out[idx] = f32_to_bf16_rne(rn);
            }
        }
    }
}

// ---------------------------------------------------------------------------
extern "C" void kernel_launch(void* const* d_in, const int* in_sizes, int n_in,
                              void* d_out, int out_size, void* d_ws, size_t ws_size,
                              hipStream_t stream) {
    (void)in_sizes; (void)n_in; (void)out_size; (void)ws_size;

    const float*         inputs = (const float*)d_in[0];  // (T, B, N) f32
    const float*         noise  = (const float*)d_in[1];  // (T, B, N) f32
    const float*         W_rec  = (const float*)d_in[2];  // (N, N)    f32
    const float*         bias   = (const float*)d_in[3];  // (1, N)    f32
    const unsigned char* mask   = (const unsigned char*)d_in[4];  // (N, N) bool

    float* out = (float*)d_out;                           // (T+1, B, N) f32

    char* ws = (char*)d_ws;
    float*          xst  = (float*)ws;                                   // 1.0 MB
    unsigned short* rbf0 = (unsigned short*)(ws + (1u << 20));           // 0.5 MB
    unsigned short* rbf1 = (unsigned short*)(ws + (1u << 20) + (512u << 10)); // 0.5 MB
    unsigned short* wbf  = (unsigned short*)(ws + (2u << 20));           // 2.0 MB

    rnn_init<<<(NBATCH * NN) / 256, 256, 0, stream>>>(xst, rbf0, out);
    rnn_mask_w<<<(NN * NN) / 256, 256, 0, stream>>>(W_rec, mask, wbf);

    const dim3 grid(NN / 128, NBATCH / 32);  // (8, 8): 64 WGs x 4 waves
    const size_t stepElems = (size_t)NBATCH * NN;
    for (int t = 0; t < TSTEPS; ++t) {
        const unsigned short* rin  = (t & 1) ? rbf1 : rbf0;
        unsigned short*       rout = (t & 1) ? rbf0 : rbf1;
        rnn_step<<<grid, 128, 0, stream>>>(
            rin, wbf,
            inputs + (size_t)t * stepElems,
            noise  + (size_t)t * stepElems,
            bias, xst,
            out + (size_t)(t + 1) * stepElems,
            rout);
    }
}
